// SelfAttentionEmbedding_12120397709753
// MI455X (gfx1250) — compile-verified, matrix operand order checked by reading, not run on hardware
//
#include <hip/hip_runtime.h>
#include <hip/hip_bf16.h>

// ---------------- problem constants ----------------
#define BB 32
#define LL 2048
#define DD 1024
#define HH 256
#define NEGV (-1.0e9f)

typedef __attribute__((ext_vector_type(16))) _Float16 v16h;
typedef __attribute__((ext_vector_type(8)))  _Float16 v8h;
typedef __attribute__((ext_vector_type(2)))  _Float16 v2h;
typedef __attribute__((ext_vector_type(8)))  float    v8f;

union V16U { v16h v; v8h h[2]; };

__device__ inline v16h mk16(v8h lo, v8h hi) {
  V16U u; u.h[0] = lo; u.h[1] = hi; return u.v;
}

__device__ inline v8f wmma32(v16h a, v16h b, v8f c) {
  // D = A(16x32 f16) * B(32x16 f16) + C(16x16 f32)
  return __builtin_amdgcn_wmma_f32_16x16x32_f16(
      /*neg_a=*/false, a, /*neg_b=*/false, b,
      /*c_mod=*/(short)0, c, /*reuse_a=*/false, /*reuse_b=*/false);
}

// =====================================================================
// Kernel 0: one-shot weight conversion to f16, transposed to B-operand
// (K-contiguous) order.  W1c[n][h][d] = W1[n][d][h]; W2c[n][d][h] = W2[n][h][d].
// 3 MB total -> stays L2-resident for all GEMM blocks.
// =====================================================================
__global__ __launch_bounds__(256) void convert_weights_kernel(
    const float* __restrict__ W1, const float* __restrict__ W2,
    _Float16* __restrict__ W1c, _Float16* __restrict__ W2c)
{
  int i = blockIdx.x * blockDim.x + threadIdx.x;   // over 3*H*D
  if (i >= 3 * HH * DD) return;
  int n = i / (HH * DD);
  int rem = i - n * (HH * DD);
  int h = rem / DD;
  int d = rem - h * DD;
  // W1c index == i == n*(H*D) + h*D + d
  W1c[i] = (_Float16)W1[(size_t)n * DD * HH + (size_t)d * HH + h];
  W2c[(size_t)n * DD * HH + (size_t)d * HH + h] =
      (_Float16)W2[(size_t)n * HH * DD + (size_t)h * DD + d];
}

// =====================================================================
// Kernel 1: fused QKV MLP.
//   h = leakyrelu(x @ W1 + b1) * bnscale + bnshift  (eval BN)
//   qkv = h @ W2 + b2   (q pre-scaled by 1/32; v written TRANSPOSED)
// Per block: one (n, b, 64-row tile).  256 threads = 8 waves (4x2).
// B operands come straight from L2-resident f16 weights (global b128).
// =====================================================================
__global__ __launch_bounds__(256) void mlp_qkv_kernel(
    const float* __restrict__ x,
    const _Float16* __restrict__ W1c, const float* __restrict__ b1,
    const float* __restrict__ gamma,  const float* __restrict__ beta,
    const float* __restrict__ rmean,  const float* __restrict__ rvar,
    const _Float16* __restrict__ W2c, const float* __restrict__ b2,
    _Float16* __restrict__ qkv16)
{
  __shared__ __align__(16) _Float16 As[64 * 128];    // x chunk [row][k]  16 KB
  __shared__ __align__(16) _Float16 Hs[64 * 256];    // hidden tile       32 KB

  const int qt = blockIdx.x;            // 0..31 row tile of L
  const int b  = blockIdx.y;            // batch
  const int n  = blockIdx.z;            // 0..2 (q,k,v)
  const int t    = threadIdx.x;
  const int w    = t >> 5;
  const int lane = t & 31;
  const int wm   = w >> 1;              // 0..3 (M tiles)
  const int wn   = w & 1;               // 0..1 (N halves)
  const int lrow = lane & 15;
  const int half = lane >> 4;           // 0/1
  const int kb   = half * 8;
  const int m0g  = qt * 64;

  const float*    xb   = x   + (size_t)b * LL * DD;
  const _Float16* W1cn = W1c + (size_t)n * HH * DD;  // [h][d]
  const _Float16* W2cn = W2c + (size_t)n * DD * HH;  // [d][h]

  const v8f vzero = {0.f,0.f,0.f,0.f,0.f,0.f,0.f,0.f};

  // ---------------- stage 1: h = act(x @ W1) ----------------
  v8f acc[8];
  #pragma unroll
  for (int i = 0; i < 8; ++i) acc[i] = vzero;

  for (int kk = 0; kk < DD; kk += 128) {
    // stage x chunk 64x128 -> As (f16), float2 loads + packed b32 stores
    #pragma unroll
    for (int i = 0; i < 16; ++i) {
      int p   = t + 256 * i;          // pair index 0..4095
      int row = p >> 6;               // 0..63
      int k   = (p & 63) * 2;         // even 0..126
      float2 xv = *(const float2*)(xb + (size_t)(m0g + row) * DD + kk + k);
      v2h hv; hv[0] = (_Float16)xv.x; hv[1] = (_Float16)xv.y;
      *(v2h*)&As[row * 128 + k] = hv;
    }
    if (kk + 128 < DD && t < 64)
      __builtin_prefetch(xb + (size_t)(m0g + t) * DD + kk + 128, 0, 0);
    __syncthreads();

    #pragma unroll
    for (int ks = 0; ks < 4; ++ks) {
      int kg = kk + ks * 32;
      v8h alo = *(const v8h*)&As[(wm * 16 + lrow) * 128 + ks * 32 + kb];
      v8h ahi = *(const v8h*)&As[(wm * 16 + lrow) * 128 + ks * 32 + kb + 16];
      v16h af = mk16(alo, ahi);
      #pragma unroll
      for (int tt = 0; tt < 8; ++tt) {
        int col = wn * 128 + tt * 16 + lrow;
        const _Float16* bp = W1cn + (size_t)col * DD + kg;
        acc[tt] = wmma32(af, mk16(*(const v8h*)(bp + kb),
                                  *(const v8h*)(bp + kb + 16)), acc[tt]);
      }
    }
    __syncthreads();
  }

  // epilogue: bias + LeakyReLU + BN affine -> Hs (f16)
  #pragma unroll
  for (int tt = 0; tt < 8; ++tt) {
    int col = wn * 128 + tt * 16 + lrow;
    float bias = b1[n * HH + col];
    float sc = gamma[n * HH + col] * rsqrtf(rvar[n * HH + col] + 1e-5f);
    float sh = beta[n * HH + col] - rmean[n * HH + col] * sc;
    #pragma unroll
    for (int r = 0; r < 8; ++r) {
      int row = wm * 16 + r + half * 8;
      float hv = acc[tt][r] + bias;
      hv = hv > 0.f ? hv : 0.01f * hv;
      hv = hv * sc + sh;
      Hs[row * 256 + col] = (_Float16)hv;
    }
  }
  __syncthreads();

  // ---------------- stage 2: qkv = h @ W2 + b2 ----------------
  _Float16* outn = qkv16 + ((size_t)n * BB + b) * (size_t)LL * DD;
  const float qscale = (n == 0) ? (1.0f / 32.0f) : 1.0f;

  for (int c = 0; c < 8; ++c) {
    int d0 = c * 128;
    v8f a2[4];
    #pragma unroll
    for (int i = 0; i < 4; ++i) a2[i] = vzero;

    #pragma unroll 2
    for (int kkh = 0; kkh < HH; kkh += 32) {
      v8h alo = *(const v8h*)&Hs[(wm * 16 + lrow) * 256 + kkh + kb];
      v8h ahi = *(const v8h*)&Hs[(wm * 16 + lrow) * 256 + kkh + kb + 16];
      v16h af = mk16(alo, ahi);
      #pragma unroll
      for (int tt = 0; tt < 4; ++tt) {
        int colg = d0 + wn * 64 + tt * 16 + lrow;
        const _Float16* bp = W2cn + (size_t)colg * HH + kkh;
        a2[tt] = wmma32(af, mk16(*(const v8h*)(bp + kb),
                                 *(const v8h*)(bp + kb + 16)), a2[tt]);
      }
    }

    #pragma unroll
    for (int tt = 0; tt < 4; ++tt) {
      int colg = d0 + wn * 64 + tt * 16 + lrow;
      float bias = b2[n * DD + colg];
      #pragma unroll
      for (int r = 0; r < 8; ++r) {
        int rowg = m0g + wm * 16 + r + half * 8;
        float v = (a2[tt][r] + bias) * qscale;
        if (n == 2)   // V stored transposed [d][l] for fast PV B-fragments
          outn[(size_t)colg * LL + rowg] = (_Float16)v;
        else
          outn[(size_t)rowg * DD + colg] = (_Float16)v;
      }
    }
  }
}

// =====================================================================
// Kernel 2: masked flash attention + masked mean pooling.
// Block: 16 query rows, 256 threads = 8 waves.
// Score GEMM: waves split the D=1024 contraction (128 each), partials
// reduced in LDS.  Each wave owns a 128-wide ctx slice (8 accumulators).
// V is pre-transposed, so PV B-fragments are contiguous b128 loads.
// =====================================================================
__global__ __launch_bounds__(256) void attn_pool_kernel(
    const _Float16* __restrict__ qkv16, const int* __restrict__ lengths,
    float* __restrict__ out)
{
  __shared__ __align__(16) float    Sp[8 * 16 * 64];   // partial scores, 32 KB
  __shared__ __align__(16) _Float16 Pls[16 * 64];      // probabilities, 2 KB
  __shared__ float rowM[16], rowL[16], rowAlpha[16];

  const int qt = blockIdx.x;          // 0..127, 16 query rows each
  const int b  = blockIdx.y;
  const int q0 = qt * 16;
  const int len = lengths[b];
  if (q0 >= len) return;              // fully masked query tile contributes 0

  const int t    = threadIdx.x;
  const int w    = t >> 5;
  const int lane = t & 31;
  const int lrow = lane & 15;
  const int half = lane >> 4;
  const int kb   = half * 8;
  const int d0w  = w * 128;

  const _Float16* q16 = qkv16 + ((size_t)0 * BB + b) * (size_t)LL * DD;
  const _Float16* k16 = qkv16 + ((size_t)1 * BB + b) * (size_t)LL * DD;
  const _Float16* vT  = qkv16 + ((size_t)2 * BB + b) * (size_t)LL * DD; // [d][l]

  if (t < 16) { rowM[t] = -3.0e38f; rowL[t] = 0.f; }
  __syncthreads();

  const v8f vzero = {0.f,0.f,0.f,0.f,0.f,0.f,0.f,0.f};
  v8f acc[8];
  #pragma unroll
  for (int i = 0; i < 8; ++i) acc[i] = vzero;

  // q fragments are key-tile invariant: hoist (wave covers K in [w*128,+128))
  v16h qf[4];
  #pragma unroll
  for (int s = 0; s < 4; ++s) {
    const _Float16* qp = q16 + (size_t)(q0 + lrow) * DD + w * 128 + s * 32;
    qf[s] = mk16(*(const v8h*)(qp + kb), *(const v8h*)(qp + kb + 16));
  }

  for (int j0 = 0; j0 < len; j0 += 64) {
    // ---- partial scores S = (q/32) . k over this wave's K slice ----
    v8f sacc[4];
    #pragma unroll
    for (int i = 0; i < 4; ++i) sacc[i] = vzero;

    #pragma unroll
    for (int s = 0; s < 4; ++s) {
      int kk = w * 128 + s * 32;
      #pragma unroll
      for (int tt = 0; tt < 4; ++tt) {
        const _Float16* krow = k16 + (size_t)(j0 + tt * 16 + lrow) * DD + kk;
        sacc[tt] = wmma32(qf[s], mk16(*(const v8h*)(krow + kb),
                                      *(const v8h*)(krow + kb + 16)), sacc[tt]);
      }
    }
    #pragma unroll
    for (int tt = 0; tt < 4; ++tt)
      #pragma unroll
      for (int r = 0; r < 8; ++r)
        Sp[(w * 16 + r + half * 8) * 64 + tt * 16 + lrow] = sacc[tt][r];
    __syncthreads();

    // ---- reduce partials + masked online softmax (one row per thread) ----
    if (t < 16) {
      float mprev = rowM[t];
      float m_new = mprev;
      for (int c2 = 0; c2 < 64; ++c2) {
        float v = 0.f;
        for (int ww = 0; ww < 8; ++ww) v += Sp[(ww * 16 + t) * 64 + c2];
        if (j0 + c2 >= len) v = NEGV;
        Sp[t * 64 + c2] = v;            // stash reduced row (read-before-write)
        m_new = fmaxf(m_new, v);
      }
      float alpha = __expf(mprev - m_new);
      float lsum = 0.f;
      for (int c2 = 0; c2 < 64; ++c2) {
        float p = __expf(Sp[t * 64 + c2] - m_new);
        lsum += p;
        Pls[t * 64 + c2] = (_Float16)p;
      }
      rowL[t] = rowL[t] * alpha + lsum;
      rowM[t] = m_new;
      rowAlpha[t] = alpha;
    }
    __syncthreads();

    // ---- ctx = ctx*alpha + P @ V (each wave: its 128-col slice) ----
    float al[8];
    #pragma unroll
    for (int r = 0; r < 8; ++r) al[r] = rowAlpha[r + half * 8];
    #pragma unroll
    for (int tt = 0; tt < 8; ++tt)
      #pragma unroll
      for (int r = 0; r < 8; ++r) acc[tt][r] *= al[r];

    v16h pf[2];
    #pragma unroll
    for (int s = 0; s < 2; ++s) {
      v8h plo = *(const v8h*)&Pls[lrow * 64 + s * 32 + kb];
      v8h phi = *(const v8h*)&Pls[lrow * 64 + s * 32 + kb + 16];
      pf[s] = mk16(plo, phi);
    }
    #pragma unroll
    for (int tt = 0; tt < 8; ++tt) {
      int colg = d0w + tt * 16 + lrow;
      #pragma unroll
      for (int s = 0; s < 2; ++s) {
        const _Float16* vp = vT + (size_t)colg * LL + j0 + s * 32;
        acc[tt] = wmma32(pf[s], mk16(*(const v8h*)(vp + kb),
                                     *(const v8h*)(vp + kb + 16)), acc[tt]);
      }
    }
    __syncthreads();
  }

  // ---- pooled += sum over valid rows of ctx_row / (rowL * len) ----
  const float inv_len = 1.0f / (float)len;
  float wr[8];
  #pragma unroll
  for (int r = 0; r < 8; ++r) {
    int rowg = q0 + r + half * 8;
    wr[r] = (rowg < len) ? inv_len / rowL[r + half * 8] : 0.f;
  }
  float* red = Sp;                      // reuse (all waves past last sync)
  #pragma unroll
  for (int tt = 0; tt < 8; ++tt) {
    float part = 0.f;
    #pragma unroll
    for (int r = 0; r < 8; ++r) part += acc[tt][r] * wr[r];
    red[(w * 8 + tt) * 32 + lane] = part;
  }
  __syncthreads();
  if (lane < 16) {
    #pragma unroll
    for (int tt = 0; tt < 8; ++tt) {
      float v = red[(w * 8 + tt) * 32 + lane] + red[(w * 8 + tt) * 32 + lane + 16];
      atomicAdd(&out[(size_t)b * DD + d0w + tt * 16 + lane], v);
    }
  }
}

__global__ void zero_out_kernel(float* __restrict__ out, int nelem) {
  int i = blockIdx.x * blockDim.x + threadIdx.x;
  if (i < nelem) out[i] = 0.f;
}

// =====================================================================
extern "C" void kernel_launch(void* const* d_in, const int* in_sizes, int n_in,
                              void* d_out, int out_size, void* d_ws, size_t ws_size,
                              hipStream_t stream) {
  const float* x      = (const float*)d_in[0];
  const int*   lens   = (const int*)  d_in[1];
  const float* W1     = (const float*)d_in[2];
  const float* b1     = (const float*)d_in[3];
  const float* gammap = (const float*)d_in[4];
  const float* betap  = (const float*)d_in[5];
  const float* rmean  = (const float*)d_in[6];
  const float* rvar   = (const float*)d_in[7];
  const float* W2     = (const float*)d_in[8];
  const float* b2     = (const float*)d_in[9];
  float* out = (float*)d_out;

  // workspace layout: [W1c f16 | W2c f16 | qkv f16]
  _Float16* W1c   = (_Float16*)d_ws;                 // 3*H*D
  _Float16* W2c   = W1c + (size_t)3 * HH * DD;       // 3*D*H
  _Float16* qkv16 = W2c + (size_t)3 * HH * DD;       // 3*B*L*D (v transposed)

  convert_weights_kernel<<<(3 * HH * DD + 255) / 256, 256, 0, stream>>>(
      W1, W2, W1c, W2c);

  dim3 g1(LL / 64, BB, 3);
  mlp_qkv_kernel<<<g1, 256, 0, stream>>>(x, W1c, b1, gammap, betap, rmean, rvar,
                                         W2c, b2, qkv16);

  zero_out_kernel<<<(BB * DD + 255) / 256, 256, 0, stream>>>(out, BB * DD);

  dim3 g2(LL / 16, BB);
  attn_pool_kernel<<<g2, 256, 0, stream>>>(qkv16, lens, out);
}
